// SelfAttentionBlock_26328149524544
// MI455X (gfx1250) — matrix operand order repaired and checked by reading, not compiled, not run on hardware
//
#include <hip/hip_runtime.h>
#include <hip/hip_bf16.h>

// ---------------------------------------------------------------------------
// Problem constants (from reference): x [2,64,48,48], 8 heads x 32 feats.
#define L_SP 2304          // 48*48
#define NB   2             // batch
#define DD   64            // in channels
#define HH   8             // heads
#define FF   32            // features per head
#define MM   256           // HH*FF, MLP width
#define NL   4608          // NB*L_SP

typedef __attribute__((ext_vector_type(16))) _Float16       v16h;
typedef __attribute__((ext_vector_type(8)))  float          v8f;
typedef __attribute__((ext_vector_type(8)))  unsigned short v8us;

union V16H { v16h v; v8us u[2]; };
union H8   { _Float16 h[8]; v8us u; };

// ---------------------------------------------------------------------------
// WMMA helper (CDNA5 wave32, v_wmma_f32_16x16x32_f16)
__device__ __forceinline__ v8f wmma16(v16h a, v16h b, v8f c) {
  return __builtin_amdgcn_wmma_f32_16x16x32_f16(
      /*neg_a=*/false, a, /*neg_b=*/false, b,
      /*c_mod=*/(short)0, c, /*reuse_a=*/false, /*reuse_b=*/false);
}

// ---------------------------------------------------------------------------
// CDNA5 async global->LDS copies (ASYNCcnt tracked; see ISA 15.18.3 op 97/98).
// Per-lane: LDS[vdst] = MEM[vaddr] for 8/16 bytes. LDS address = low 32 bits
// of the generic pointer (ISA 10.2: LDS aperture truncates to addr[31:0]).
__device__ __forceinline__ void async_ld_b128(const _Float16* g, _Float16* l) {
  unsigned lo = (unsigned)(size_t)(void*)l;
  asm volatile("global_load_async_to_lds_b128 %0, %1, off"
               :: "v"(lo), "v"(g) : "memory");
}
__device__ __forceinline__ void async_ld_b64(const _Float16* g, _Float16* l) {
  unsigned lo = (unsigned)(size_t)(void*)l;
  asm volatile("global_load_async_to_lds_b64 %0, %1, off"
               :: "v"(lo), "v"(g) : "memory");
}
__device__ __forceinline__ void wait_async0() {
  asm volatile("s_wait_asynccnt 0x0" ::: "memory");
}

// ---------------------------------------------------------------------------
// A fragment: 16x32 f16, row-major source, leading dim ld (halves).
// Per-lane data = two contiguous 16B runs -> two b128 loads.
__device__ __forceinline__ v16h frag_load_a(const _Float16* __restrict__ src, int ld) {
  const int lane = threadIdx.x & 31;
  const _Float16* p = src + (lane & 15) * ld + ((lane & 16) ? 8 : 0);
  V16H r;
  r.u[0] = *(const v8us*)(p);
  r.u[1] = *(const v8us*)(p + 16);
  return r.v;
}

// B fragment B[k][n] built from a TRANSPOSED source: src row-major [n][k].
__device__ __forceinline__ v16h frag_load_bt(const _Float16* __restrict__ src, int ld) {
  const int lane = threadIdx.x & 31;
  const _Float16* p = src + (lane & 15) * ld + ((lane & 16) ? 8 : 0);
  V16H r;
  r.u[0] = *(const v8us*)(p);
  r.u[1] = *(const v8us*)(p + 16);
  return r.v;
}
// C/D element (VGPR i): row = i + (lane>=16 ? 8 : 0), col = lane & 15.

// ---------------------------------------------------------------------------
// K0a: NCHW -> [n*L+l, d] f16
__global__ void k_pack_x(const float* __restrict__ x, _Float16* __restrict__ flat16) {
  int i = blockIdx.x * blockDim.x + threadIdx.x;   // NL*DD threads
  if (i >= NL * DD) return;
  int d = i & (DD - 1);
  int r = i >> 6;                 // n*L + l
  int n = r / L_SP;
  int l = r - n * L_SP;
  flat16[i] = (_Float16)x[((size_t)n * DD + d) * L_SP + l];
}

// K0b: Wq/Wk/Wv [h][d][f] f32 -> [h][f][d] f16 (transposed for bt loads)
__global__ void k_conv_wqkv_t(const float* __restrict__ src, _Float16* __restrict__ dst) {
  int i = blockIdx.x * blockDim.x + threadIdx.x;   // HH*DD*FF
  if (i >= HH * DD * FF) return;
  int h = i / (DD * FF);
  int rem = i - h * DD * FF;
  int d = rem / FF;
  int f = rem - d * FF;
  dst[(h * FF + f) * DD + d] = (_Float16)src[i];
}

// K0c: W [in][out] f32 -> [out][in] f16
__global__ void k_conv_w_t(const float* __restrict__ src, _Float16* __restrict__ dst) {
  int i = blockIdx.x * blockDim.x + threadIdx.x;   // MM*MM
  if (i >= MM * MM) return;
  int r = i / MM, c = i - r * MM;
  dst[c * MM + r] = (_Float16)src[i];
}

// ---------------------------------------------------------------------------
// K1: per-head projection  y = relu(inorm_32(flat @ W_h + b_h))  -> f16
// Output indexing dst[l*rs + f*cs] lets the same kernel emit Q/K row-major
// ([hn][l][f]) and V transposed ([hn][f][l]).
// grid (NL/64, HH), block 128 (4 waves, 16 rows each).
__global__ __launch_bounds__(128) void k_proj(
    const _Float16* __restrict__ flat16,   // [NL, 64]
    const _Float16* __restrict__ Wt16,     // [HH, 32, 64] (f-major, transposed)
    const float*    __restrict__ bias,     // [HH, 32]
    _Float16*       __restrict__ out,      // [hn] base + l*rs + f*cs
    int rs, int cs)
{
  __shared__ _Float16 sWt[FF * DD];        // [f][d]
  __shared__ float    sBias[FF];
  __shared__ float    sC[4][16][FF];

  const int h     = blockIdx.y;
  const int rbase = blockIdx.x * 64;
  const int tid   = threadIdx.x;
  {
    const unsigned* wsrc = (const unsigned*)(Wt16 + h * FF * DD);
    unsigned* wdst = (unsigned*)sWt;
    for (int i = tid; i < FF * DD / 2; i += 128) wdst[i] = wsrc[i];
    if (tid < FF) sBias[tid] = bias[h * FF + tid];
  }
  __syncthreads();

  const int w    = tid >> 5;
  const int lane = tid & 31;
  const int rw   = rbase + w * 16;

  v8f c0 = {}, c1 = {};
#pragma unroll
  for (int kk = 0; kk < DD; kk += 32) {
    v16h a  = frag_load_a(flat16 + (size_t)rw * DD + kk, DD);
    v16h b0 = frag_load_bt(sWt + kk, DD);            // cols f 0..15
    v16h b1 = frag_load_bt(sWt + 16 * DD + kk, DD);  // cols f 16..31
    c0 = wmma16(a, b0, c0);
    c1 = wmma16(a, b1, c1);
  }
  {
    const int col = lane & 15;
    const int ro  = (lane & 16) ? 8 : 0;
#pragma unroll
    for (int i = 0; i < 8; ++i) {
      sC[w][ro + i][col]      = c0[i] + sBias[col];
      sC[w][ro + i][col + 16] = c1[i] + sBias[col + 16];
    }
  }
  __syncthreads();
  if (tid < 64) {                      // one thread per row: inorm(32) + relu
    int wv = tid >> 4, rl = tid & 15;
    float s = 0.f, s2 = 0.f;
#pragma unroll
    for (int f = 0; f < FF; ++f) { float v = sC[wv][rl][f]; s += v; s2 += v * v; }
    float mean = s * (1.0f / FF);
    float var  = s2 * (1.0f / FF) - mean * mean;
    float rstd = rsqrtf(var + 1e-5f);
    int r = rbase + wv * 16 + rl;      // n*L + l
    int n = r / L_SP;
    int l = r - n * L_SP;
    _Float16* dst = out + ((size_t)h * NB + n) * L_SP * FF;
#pragma unroll
    for (int f = 0; f < FF; ++f) {
      float v = (sC[wv][rl][f] - mean) * rstd;
      dst[l * rs + f * cs] = (_Float16)fmaxf(v, 0.f);
    }
  }
}

// ---------------------------------------------------------------------------
// K2: column-softmax stats. T[m,l] = (K Q^T)[m,l]/sqrt(F); per-row (m) online
// max/sum over all l. Q tiles streamed into LDS with async copies (double
// buffered, shared by all 4 waves).  grid (L/64, HH*NB), block 128.
__global__ __launch_bounds__(128) void k_stats(
    const _Float16* __restrict__ Q16, const _Float16* __restrict__ K16,
    float* __restrict__ rmax, float* __restrict__ rsum)
{
  __shared__ _Float16 sQ[2][16 * FF];          // 2 x 1KB
  const int hn   = blockIdx.y;
  const int tid  = threadIdx.x;
  const int w    = tid >> 5, lane = tid & 31;
  const int mb   = blockIdx.x * 64 + w * 16;
  const _Float16* Kb = K16 + (size_t)hn * L_SP * FF;
  const _Float16* Qb = Q16 + (size_t)hn * L_SP * FF;
  const float scale = 0.17677669529663687f;    // 1/sqrt(32)

  auto stage = [&](int lt, int buf) {          // 128 threads x 8 bytes
    const int row = tid >> 3, seg = tid & 7;
    async_ld_b64(Qb + (size_t)(lt * 16 + row) * FF + seg * 4,
                 &sQ[buf][row * FF + seg * 4]);
  };

  v16h a = frag_load_a(Kb + (size_t)mb * FF, FF);   // K rows fixed for this wave
  float pmax[8], psum[8];
#pragma unroll
  for (int i = 0; i < 8; ++i) { pmax[i] = -3.0e38f; psum[i] = 0.f; }

  stage(0, 0);
  wait_async0();
  __syncthreads();

  const int NT = L_SP / 16;
  for (int lt = 0; lt < NT; ++lt) {
    const int cur = lt & 1;
    if (lt + 1 < NT) stage(lt + 1, cur ^ 1);
    v16h b = frag_load_bt(sQ[cur], FF);
    v8f c = {};
    c = wmma16(a, b, c);
#pragma unroll
    for (int i = 0; i < 8; ++i) {
      float v = c[i] * scale;
      if (v > pmax[i]) {
        psum[i] = psum[i] * __expf(pmax[i] - v) + 1.0f;
        pmax[i] = v;
      } else {
        psum[i] += __expf(v - pmax[i]);
      }
    }
    wait_async0();
    __syncthreads();
  }
  // merge (max,sum) across the 16 lanes of each half-wave (C-layout columns)
#pragma unroll
  for (int i = 0; i < 8; ++i) {
    float m = pmax[i], s = psum[i];
#pragma unroll
    for (int d = 1; d < 16; d <<= 1) {
      float om = __shfl_xor(m, d, 32);
      float os = __shfl_xor(s, d, 32);
      float nm = fmaxf(m, om);
      s = s * __expf(m - nm) + os * __expf(om - nm);
      m = nm;
    }
    pmax[i] = m; psum[i] = s;
  }
  if ((lane & 15) == 0) {
    const int ro = (lane & 16) ? 8 : 0;
#pragma unroll
    for (int i = 0; i < 8; ++i) {
      int mr = mb + ro + i;
      rmax[(size_t)hn * L_SP + mr] = pmax[i];
      rsum[(size_t)hn * L_SP + mr] = psum[i];
    }
  }
}

// ---------------------------------------------------------------------------
// K3: A[l,f] = sum_m P[m,l] V[m,f], P = exp(T - rmax)/rsum. K/V tiles streamed
// into LDS with async copies (double buffered, shared by all 4 waves); P goes
// through LDS (packed b128) to become the next WMMA A fragment.
// grid (L/64, HH*NB), block 128.
__global__ __launch_bounds__(128) void k_att(
    const _Float16* __restrict__ Q16, const _Float16* __restrict__ K16,
    const _Float16* __restrict__ V16t,   // [hn][f][l]  (transposed)
    const float* __restrict__ rmax, const float* __restrict__ rsum,
    float* __restrict__ heads32, _Float16* __restrict__ heads16)
{
  __shared__ _Float16 sP[4][16][32];   // per-wave P^T tile: [l_local][m_local]
  __shared__ _Float16 sK[2][32 * FF];  // [buf][m(32)][f(32)]
  __shared__ _Float16 sV[2][FF * 32];  // [buf][f(32)][m(32)]
  __shared__ float sRmax[L_SP];
  __shared__ float sRinv[L_SP];
  const int hn  = blockIdx.y;
  const int h   = hn >> 1, n = hn & 1;
  const int tid = threadIdx.x;
  const int w   = tid >> 5, lane = tid & 31;
  const int lb  = blockIdx.x * 64 + w * 16;
  const _Float16* Kb = K16  + (size_t)hn * L_SP * FF;
  const _Float16* Qb = Q16  + (size_t)hn * L_SP * FF;
  const _Float16* Vt = V16t + (size_t)hn * L_SP * FF;   // [f][l]
  const float scale = 0.17677669529663687f;

  auto stage = [&](int mt, int buf) {          // 128 threads x 16B per tile
    const int row = tid >> 2, seg = tid & 3;   // row: m-row for K, f-row for V
    const int mb = mt * 32;
    async_ld_b128(Kb + (size_t)(mb + row) * FF + seg * 8,
                  &sK[buf][row * FF + seg * 8]);
    async_ld_b128(Vt + (size_t)row * L_SP + mb + seg * 8,
                  &sV[buf][row * 32 + seg * 8]);
  };

  {
    const float* rmx = rmax + (size_t)hn * L_SP;
    const float* rsm = rsum + (size_t)hn * L_SP;
    for (int i = tid; i < L_SP; i += 128) {
      sRmax[i] = rmx[i];
      sRinv[i] = 1.0f / rsm[i];
    }
  }

  v16h bq = frag_load_bt(Qb + (size_t)lb * FF, FF);  // fixed Q cols for this wave
  const int col = lane & 15;
  const int ro  = (lane & 16) ? 8 : 0;

  stage(0, 0);
  wait_async0();
  __syncthreads();

  v8f c0 = {}, c1 = {};
  const int NT = L_SP / 32;
  for (int mt = 0; mt < NT; ++mt) {
    const int cur = mt & 1;
    const int mb = mt * 32;
    if (mt + 1 < NT) stage(mt + 1, cur ^ 1);
#pragma unroll
    for (int s = 0; s < 2; ++s) {
      v16h ak = frag_load_a(&sK[cur][s * 16 * FF], FF);
      v8f t = {};
      t = wmma16(ak, bq, t);
      H8 t8;
#pragma unroll
      for (int i = 0; i < 8; ++i) {
        int mr = mb + s * 16 + ro + i;
        t8.h[i] = (_Float16)(__expf(t[i] * scale - sRmax[mr]) * sRinv[mr]);
      }
      *(v8us*)&sP[w][col][s * 16 + ro] = t8.u;       // one b128 store
    }
    __syncthreads();                                 // publish P^T
    v16h ap  = frag_load_a(&sP[w][0][0], 32);        // ds_load_b128 x2
    v16h bv0 = frag_load_bt(&sV[cur][0], 32);        // f 0..15
    v16h bv1 = frag_load_bt(&sV[cur][16 * 32], 32);  // f 16..31
    c0 = wmma16(ap, bv0, c0);
    c1 = wmma16(ap, bv1, c1);
    wait_async0();
    __syncthreads();                                 // next K/V buffer ready
  }
#pragma unroll
  for (int i = 0; i < 8; ++i) {
    int l = lb + ro + i;
    size_t base = ((size_t)n * L_SP + l) * MM + h * FF;
    heads32[base + col]      = c0[i];
    heads32[base + col + 16] = c1[i];
    heads16[base + col]      = (_Float16)c0[i];
    heads16[base + col + 16] = (_Float16)c1[i];
  }
}

// ---------------------------------------------------------------------------
// K4: m1 = relu(heads @ W1 + b1), f16 out. grid (NL/64, MM/32), block 128.
__global__ __launch_bounds__(128) void k_mlp1(
    const _Float16* __restrict__ heads16, const _Float16* __restrict__ W1t,  // [out][in]
    const float* __restrict__ bias1, _Float16* __restrict__ m1_16)
{
  const int tid  = threadIdx.x;
  const int w    = tid >> 5, lane = tid & 31;
  const int rb   = blockIdx.x * 64 + w * 16;
  const int cb   = blockIdx.y * 32;
  v8f c0 = {}, c1 = {};
#pragma unroll
  for (int kk = 0; kk < MM; kk += 32) {
    v16h a  = frag_load_a(heads16 + (size_t)rb * MM + kk, MM);
    v16h b0 = frag_load_bt(W1t + (size_t)cb * MM + kk, MM);
    v16h b1 = frag_load_bt(W1t + (size_t)(cb + 16) * MM + kk, MM);
    c0 = wmma16(a, b0, c0);
    c1 = wmma16(a, b1, c1);
  }
  const int col = lane & 15;
  const int ro  = (lane & 16) ? 8 : 0;
  float bb0 = bias1[cb + col], bb1 = bias1[cb + col + 16];
#pragma unroll
  for (int i = 0; i < 8; ++i) {
    size_t base = (size_t)(rb + ro + i) * MM + cb;
    m1_16[base + col]      = (_Float16)fmaxf(c0[i] + bb0, 0.f);
    m1_16[base + col + 16] = (_Float16)fmaxf(c1[i] + bb1, 0.f);
  }
}

// ---------------------------------------------------------------------------
// K5: m2 = relu(m1 @ W2 + b2); o = inorm_256(m2 + heads)*gamma + beta;
// transpose to [N, 256, H, W]. grid NL/16, block 256 (8 waves cover 256 cols).
__global__ __launch_bounds__(256) void k_mlp2norm(
    const _Float16* __restrict__ m1_16, const _Float16* __restrict__ W2t,  // [out][in]
    const float* __restrict__ bias2, const float* __restrict__ heads32,
    const float* __restrict__ gamma, const float* __restrict__ beta,
    float* __restrict__ out)
{
  __shared__ float sT[16][MM];        // 16 rows x 256 cols
  __shared__ float sPs[16][16];
  __shared__ float sPs2[16][16];
  __shared__ float sMean[16], sRstd[16];

  const int tid  = threadIdx.x;
  const int w    = tid >> 5, lane = tid & 31;
  const int rb   = blockIdx.x * 16;
  const int cb   = w * 32;
  v8f c0 = {}, c1 = {};
#pragma unroll
  for (int kk = 0; kk < MM; kk += 32) {
    v16h a  = frag_load_a(m1_16 + (size_t)rb * MM + kk, MM);
    v16h b0 = frag_load_bt(W2t + (size_t)cb * MM + kk, MM);
    v16h b1 = frag_load_bt(W2t + (size_t)(cb + 16) * MM + kk, MM);
    c0 = wmma16(a, b0, c0);
    c1 = wmma16(a, b1, c1);
  }
  const int col = lane & 15;
  const int ro  = (lane & 16) ? 8 : 0;
  float bb0 = bias2[cb + col], bb1 = bias2[cb + col + 16];
#pragma unroll
  for (int i = 0; i < 8; ++i) {
    size_t base = (size_t)(rb + ro + i) * MM + cb;
    float v0 = fmaxf(c0[i] + bb0, 0.f) + heads32[base + col];
    float v1 = fmaxf(c1[i] + bb1, 0.f) + heads32[base + col + 16];
    sT[ro + i][cb + col]      = v0;
    sT[ro + i][cb + col + 16] = v1;
  }
  __syncthreads();
  {
    int row = tid >> 4, seg = tid & 15;
    float s = 0.f, s2 = 0.f;
#pragma unroll
    for (int j = 0; j < 16; ++j) {
      float v = sT[row][seg * 16 + j];
      s += v; s2 += v * v;
    }
    sPs[row][seg] = s; sPs2[row][seg] = s2;
  }
  __syncthreads();
  if (tid < 16) {
    float s = 0.f, s2 = 0.f;
#pragma unroll
    for (int j = 0; j < 16; ++j) { s += sPs[tid][j]; s2 += sPs2[tid][j]; }
    float mean = s * (1.0f / MM);
    float var  = s2 * (1.0f / MM) - mean * mean;
    sMean[tid] = mean;
    sRstd[tid] = rsqrtf(var + 1e-5f);
  }
  __syncthreads();
  {
    int row = tid >> 4, seg = tid & 15;
    int r = rb + row;
    int n = r / L_SP;
    int l = r - n * L_SP;
    float mean = sMean[row], rstd = sRstd[row];
#pragma unroll
    for (int j = 0; j < 16; ++j) {
      int c = seg * 16 + j;
      float v = (sT[row][c] - mean) * rstd;
      v = v * gamma[c] + beta[c];
      out[((size_t)n * MM + c) * L_SP + l] = v;   // [N, 256, H*W]
    }
  }
}

// ---------------------------------------------------------------------------
extern "C" void kernel_launch(void* const* d_in, const int* in_sizes, int n_in,
                              void* d_out, int out_size, void* d_ws, size_t ws_size,
                              hipStream_t stream) {
  (void)in_sizes; (void)n_in; (void)out_size; (void)ws_size;
  const float* x     = (const float*)d_in[0];
  const float* Wq    = (const float*)d_in[1];
  const float* bq    = (const float*)d_in[2];
  const float* Wk    = (const float*)d_in[3];
  const float* bk    = (const float*)d_in[4];
  const float* Wv    = (const float*)d_in[5];
  const float* bv    = (const float*)d_in[6];
  const float* W1    = (const float*)d_in[7];
  const float* b1    = (const float*)d_in[8];
  const float* W2    = (const float*)d_in[9];
  const float* b2    = (const float*)d_in[10];
  const float* gamma = (const float*)d_in[11];
  const float* beta  = (const float*)d_in[12];
  float* out = (float*)d_out;

  // workspace carve-out (~18 MB total)
  char* p = (char*)d_ws;
  auto carve = [&](size_t bytes) {
    char* q = p;
    p += (bytes + 255) & ~(size_t)255;
    return q;
  };
  _Float16* flat16  = (_Float16*)carve((size_t)NL * DD * 2);
  _Float16* wqT     = (_Float16*)carve((size_t)HH * DD * FF * 2);
  _Float16* wkT     = (_Float16*)carve((size_t)HH * DD * FF * 2);
  _Float16* wvT     = (_Float16*)carve((size_t)HH * DD * FF * 2);
  _Float16* w1T     = (_Float16*)carve((size_t)MM * MM * 2);
  _Float16* w2T     = (_Float16*)carve((size_t)MM * MM * 2);
  _Float16* Q16     = (_Float16*)carve((size_t)HH * NB * L_SP * FF * 2);
  _Float16* K16     = (_Float16*)carve((size_t)HH * NB * L_SP * FF * 2);
  _Float16* V16t    = (_Float16*)carve((size_t)HH * NB * L_SP * FF * 2);
  float*    rmaxb   = (float*)carve((size_t)HH * NB * L_SP * 4);
  float*    rsumb   = (float*)carve((size_t)HH * NB * L_SP * 4);
  float*    heads32 = (float*)carve((size_t)NL * MM * 4);
  _Float16* heads16 = (_Float16*)carve((size_t)NL * MM * 2);
  _Float16* m1_16   = (_Float16*)carve((size_t)NL * MM * 2);

  // precision staging (all B operands stored transposed for b128 frag loads)
  k_pack_x<<<(NL * DD) / 256, 256, 0, stream>>>(x, flat16);
  k_conv_wqkv_t<<<(HH * DD * FF + 255) / 256, 256, 0, stream>>>(Wq, wqT);
  k_conv_wqkv_t<<<(HH * DD * FF + 255) / 256, 256, 0, stream>>>(Wk, wkT);
  k_conv_wqkv_t<<<(HH * DD * FF + 255) / 256, 256, 0, stream>>>(Wv, wvT);
  k_conv_w_t<<<(MM * MM + 255) / 256, 256, 0, stream>>>(W1, w1T);
  k_conv_w_t<<<(MM * MM + 255) / 256, 256, 0, stream>>>(W2, w2T);

  // Q/K/V projections (WMMA + inorm + relu); V emitted transposed [hn][f][l]
  dim3 gProj(NL / 64, HH);
  k_proj<<<gProj, 128, 0, stream>>>(flat16, wqT, bq, Q16, FF, 1);
  k_proj<<<gProj, 128, 0, stream>>>(flat16, wkT, bk, K16, FF, 1);
  k_proj<<<gProj, 128, 0, stream>>>(flat16, wvT, bv, V16t, 1, L_SP);

  // attention: two-pass column softmax (no score materialization)
  dim3 gAtt(L_SP / 64, HH * NB);
  k_stats<<<gAtt, 128, 0, stream>>>(Q16, K16, rmaxb, rsumb);
  k_att<<<gAtt, 128, 0, stream>>>(Q16, K16, V16t, rmaxb, rsumb, heads32, heads16);

  // MLP + residual + inorm + affine + NCHW transpose
  dim3 gM1(NL / 64, MM / 32);
  k_mlp1<<<gM1, 128, 0, stream>>>(heads16, w1T, b1, m1_16);
  k_mlp2norm<<<NL / 16, 256, 0, stream>>>(m1_16, w2T, b2, heads32, gamma, beta, out);
}
